// Attention_55679956025809
// MI455X (gfx1250) — matrix operand order repaired and checked by reading, not compiled, not run on hardware
//
#include <hip/hip_runtime.h>
#include <hip/hip_bf16.h>

// ---------------------------------------------------------------------------
// Types for WMMA fragments (wave32, gfx1250)
// ---------------------------------------------------------------------------
typedef _Float16 half4  __attribute__((ext_vector_type(4)));
typedef _Float16 half8  __attribute__((ext_vector_type(8)));
typedef _Float16 half16 __attribute__((ext_vector_type(16)));
typedef float    f32x8  __attribute__((ext_vector_type(8)));

static __device__ __forceinline__ half16 hcat(half8 a, half8 b) {
    return __builtin_shufflevector(a, b, 0,1,2,3,4,5,6,7,8,9,10,11,12,13,14,15);
}

static __device__ __forceinline__ f32x8 wmma_f16(half16 a, half16 b, f32x8 c) {
    // v_wmma_f32_16x16x32_f16  (neg_a, A, neg_b, B, c_mod, C, reuse_a, reuse_b)
    return __builtin_amdgcn_wmma_f32_16x16x32_f16(false, a, false, b, (short)0, c, false, false);
}

static __device__ __forceinline__ float fast_exp2(float x) {
    return __builtin_amdgcn_exp2f(x);   // v_exp_f32
}

// Async copy: 16 bytes global -> LDS, tracked by ASYNCcnt (CDNA5).
// Low 32 bits of a generic pointer to __shared__ are the LDS byte address.
static __device__ __forceinline__ void async_cp16(void* lds_dst, const void* gsrc) {
    unsigned loff = (unsigned)(unsigned long long)lds_dst;
    asm volatile("global_load_async_to_lds_b128 %0, %1, off"
                 :: "v"(loff), "v"(gsrc) : "memory");
}
static __device__ __forceinline__ void wait_async0() {
    asm volatile("s_wait_asynccnt 0x0" ::: "memory");
}

// Problem constants
#define BB   2
#define NN   2048
#define DD   1024
#define HH   16
#define DH   64
#define ROWS (BB*NN)          // 4096
#define SCALE_LOG2E (0.125f * 1.44269504088896f)   // DH^-0.5 * log2(e)

// ---------------------------------------------------------------------------
// 1) Weight convert + transpose: in [K, C] fp32 row-major -> out [C, K] f16
// ---------------------------------------------------------------------------
__global__ __launch_bounds__(256) void wconvT_kernel(const float* __restrict__ in,
                                                     _Float16* __restrict__ out,
                                                     int K, int C) {
    int idx = blockIdx.x * 256 + threadIdx.x;
    if (idx >= K * C) return;
    int k = idx / C;
    int c = idx - k * C;                 // reads coalesced along c
    out[(size_t)c * K + k] = (_Float16)in[idx];
}

// ---------------------------------------------------------------------------
// 2) LayerNorm: x [4096,1024] fp32 -> xn [4096,1024] f16 (biased var, eps=1e-5)
// ---------------------------------------------------------------------------
__global__ __launch_bounds__(256) void layernorm_kernel(const float* __restrict__ x,
                                                        const float* __restrict__ g,
                                                        const float* __restrict__ bvec,
                                                        _Float16* __restrict__ xn) {
    __shared__ float red[16];
    const int row = blockIdx.x;
    const int tid = threadIdx.x;
    const float4 v = ((const float4*)(x + (size_t)row * DD))[tid];

    float s  = v.x + v.y + v.z + v.w;
    float ss = v.x*v.x + v.y*v.y + v.z*v.z + v.w*v.w;
    #pragma unroll
    for (int m = 1; m < 32; m <<= 1) {
        s  += __shfl_xor(s,  m, 32);
        ss += __shfl_xor(ss, m, 32);
    }
    const int w = tid >> 5;
    if ((tid & 31) == 0) { red[w] = s; red[8 + w] = ss; }
    __syncthreads();
    float S = 0.f, SS = 0.f;
    #pragma unroll
    for (int i = 0; i < 8; ++i) { S += red[i]; SS += red[8 + i]; }
    const float mu   = S * (1.0f / DD);
    const float var  = SS * (1.0f / DD) - mu * mu;
    const float rstd = rsqrtf(var + 1e-5f);

    const float4 gg = ((const float4*)g)[tid];
    const float4 bb = ((const float4*)bvec)[tid];
    half4 o;
    o[0] = (_Float16)((v.x - mu) * rstd * gg.x + bb.x);
    o[1] = (_Float16)((v.y - mu) * rstd * gg.y + bb.y);
    o[2] = (_Float16)((v.z - mu) * rstd * gg.z + bb.z);
    o[3] = (_Float16)((v.w - mu) * rstd * gg.w + bb.w);
    *(half4*)(xn + (size_t)row * DD + tid * 4) = o;
}

// ---------------------------------------------------------------------------
// 3) QKV GEMM: xn[4096,1024] @ [Wq|Wkv] -> Q/K head-major, V transposed.
//    One wave per 32x64 tile; ping-pong 2x-unrolled K loop (no reg copies).
//    launch_bounds(256,2): allow up to 512 VGPRs so acc + both fragment
//    buffers stay resident (no accumulator shuffling).
// ---------------------------------------------------------------------------
__global__ __launch_bounds__(256, 2) void gemm_qkv_kernel(const _Float16* __restrict__ xn,
                                                          const _Float16* __restrict__ wqT,   // [1024,1024]
                                                          const _Float16* __restrict__ wkvT,  // [2048,1024]
                                                          _Float16* __restrict__ qbuf,        // [B,H,N,DH]
                                                          _Float16* __restrict__ kbuf,        // [B,H,N,DH]
                                                          _Float16* __restrict__ vtbuf) {     // [B,H,DH,N]
    const int lane = threadIdx.x & 31;
    const int wid  = blockIdx.x * 8 + (threadIdx.x >> 5);
    const int mIdx = wid & 127;          // 128 M-tiles of 32 rows
    const int cIdx = wid >> 7;           // 48 column tiles of 64 (3072 cols)
    const int lh   = lane >> 4;
    const int ln   = lane & 15;
    const int m0   = mIdx * 32;
    const int c0   = cIdx * 64;

    const _Float16* wrowp[4];
    #pragma unroll
    for (int t = 0; t < 4; ++t) {
        const int c = c0 + t * 16 + ln;
        wrowp[t] = (c < 1024) ? (wqT + (size_t)c * DD) : (wkvT + (size_t)(c - 1024) * DD);
    }
    const _Float16* arow[2] = { xn + (size_t)(m0 + ln) * DD,
                                xn + (size_t)(m0 + 16 + ln) * DD };

    f32x8 acc[2][4] = {};
    half16 aA[2], bA[4], aB[2], bB[4];

    auto loadA = [&](int k) {
        #pragma unroll
        for (int rb = 0; rb < 2; ++rb)
            aA[rb] = *(const half16*)(arow[rb] + k + lh * 16);
        #pragma unroll
        for (int t = 0; t < 4; ++t) bA[t] = *(const half16*)(wrowp[t] + k + lh * 16);
    };
    auto loadB = [&](int k) {
        #pragma unroll
        for (int rb = 0; rb < 2; ++rb)
            aB[rb] = *(const half16*)(arow[rb] + k + lh * 16);
        #pragma unroll
        for (int t = 0; t < 4; ++t) bB[t] = *(const half16*)(wrowp[t] + k + lh * 16);
    };
    auto compA = [&]() {
        #pragma unroll
        for (int rb = 0; rb < 2; ++rb)
            #pragma unroll
            for (int t = 0; t < 4; ++t) acc[rb][t] = wmma_f16(aA[rb], bA[t], acc[rb][t]);
    };
    auto compB = [&]() {
        #pragma unroll
        for (int rb = 0; rb < 2; ++rb)
            #pragma unroll
            for (int t = 0; t < 4; ++t) acc[rb][t] = wmma_f16(aB[rb], bB[t], acc[rb][t]);
    };

    loadA(0);
    for (int k = 0; k < DD - 64; k += 64) {
        loadB(k + 32);
        compA();
        loadA(k + 64);
        compB();
    }
    loadB(DD - 32);
    compA();
    compB();

    // Epilogue: scatter to head layouts as f16
    #pragma unroll
    for (int rb = 0; rb < 2; ++rb) {
        #pragma unroll
        for (int t = 0; t < 4; ++t) {
            #pragma unroll
            for (int r = 0; r < 8; ++r) {
                const int m = m0 + rb * 16 + lh * 8 + r;  // global row
                const int b = m >> 11;                    // /N
                const int n = m & (NN - 1);
                const int c = c0 + t * 16 + ln;
                const _Float16 val = (_Float16)acc[rb][t][r];
                if (c < 1024) {
                    const int h = c >> 6, d = c & 63;
                    qbuf[(((size_t)b * HH + h) * NN + n) * DH + d] = val;
                } else if (c < 2048) {
                    const int c2 = c - 1024, h = c2 >> 6, d = c2 & 63;
                    kbuf[(((size_t)b * HH + h) * NN + n) * DH + d] = val;
                } else {
                    const int c2 = c - 2048, h = c2 >> 6, d = c2 & 63;
                    vtbuf[(((size_t)b * HH + h) * DH + d) * NN + n] = val;
                }
            }
        }
    }
}

// ---------------------------------------------------------------------------
// 4) Flash attention, async-LDS staged. The 8 waves of a block share (b,h);
//    32-key K/V tiles staged via global_load_async_to_lds (ASYNCcnt),
//    double-buffered: wait stage i -> barrier -> issue stage i+1 -> compute.
// ---------------------------------------------------------------------------
__global__ __launch_bounds__(256) void attn_kernel(const _Float16* __restrict__ qbuf,
                                                   const _Float16* __restrict__ kbuf,
                                                   const _Float16* __restrict__ vtbuf,
                                                   _Float16* __restrict__ obuf) {  // [B*N, H*DH]
    __shared__ _Float16 kTile[2][32 * 64];    // [key][d]   4 KB per buffer
    __shared__ _Float16 vTile[2][64 * 32];    // [d][key]   4 KB per buffer
    __shared__ _Float16 plds[8][16 * 32];     // P transpose, 1 KB per wave

    const int tid   = threadIdx.x;
    const int lane  = tid & 31;
    const int wslot = tid >> 5;
    const int wid   = blockIdx.x * 8 + wslot;     // 4096 waves total
    const int bh    = wid >> 7;                   // b*H + h (8 waves/block share it)
    const int mtile = wid & 127;
    const int lh    = lane >> 4;
    const int ln    = lane & 15;

    const _Float16* qh = qbuf  + (size_t)bh * NN * DH;
    const _Float16* kh = kbuf  + (size_t)bh * NN * DH;
    const _Float16* vh = vtbuf + (size_t)bh * DH * NN;
    _Float16* pl = plds[wslot];

    // Per-thread async-copy slots: K tile contiguous 4KB; V tile 64 x 64B rows.
    const int vrow = tid >> 2;        // 0..63  (head dim)
    const int vchk = tid & 3;         // 0..3   (16B chunk within 32 keys)

    // Q fragments (16 rows x 64 dims = two 16x32 A fragments, kept in VGPRs)
    const _Float16* qrow = qh + (size_t)(mtile * 16 + ln) * DH;
    half16 qf[2];
    #pragma unroll
    for (int h2 = 0; h2 < 2; ++h2)
        qf[h2] = *(const half16*)(qrow + h2 * 32 + lh * 16);

    f32x8 accO[4] = {};
    float mrow[8], lrow[8];
    #pragma unroll
    for (int r = 0; r < 8; ++r) { mrow[r] = -3.0e38f; lrow[r] = 0.0f; }

    // Prologue: stage jb = 0 into buffer 0 (2 async ops per thread)
    async_cp16((char*)kTile[0] + tid * 16, (const char*)kh + tid * 16);
    async_cp16((char*)vTile[0] + vrow * 64 + vchk * 16,
               (const char*)(vh + (size_t)vrow * NN) + vchk * 16);

    for (int jb = 0; jb < NN; jb += 32) {
        const int buf = (jb >> 5) & 1;

        // Stage `buf` complete in LDS for this wave, then for all waves.
        wait_async0();
        __syncthreads();

        // Issue next stage into the other buffer (everyone finished reading it
        // before arriving at the barrier above).
        const int jn = jb + 32;
        if (jn < NN) {
            const int nb = buf ^ 1;
            async_cp16((char*)kTile[nb] + tid * 16,
                       (const char*)(kh + (size_t)jn * DH) + tid * 16);
            async_cp16((char*)vTile[nb] + vrow * 64 + vchk * 16,
                       (const char*)(vh + (size_t)vrow * NN + jn) + vchk * 16);
        }

        // ---- S = Q @ K^T for two 16x16 key tiles (4 WMMAs), K from LDS ----
        const _Float16* kt = kTile[buf];
        f32x8 s0 = {}, s1 = {};
        #pragma unroll
        for (int h2 = 0; h2 < 2; ++h2) {
            const half16 b0 = *(const half16*)(kt + ln * 64        + h2 * 32 + lh * 16);
            s0 = wmma_f16(qf[h2], b0, s0);
            const half16 b1 = *(const half16*)(kt + (16 + ln) * 64 + h2 * 32 + lh * 16);
            s1 = wmma_f16(qf[h2], b1, s1);
        }

        // ---- Online softmax (row = lh*8 + r; reductions across 16-lane half) ----
        #pragma unroll
        for (int r = 0; r < 8; ++r) {
            const float t0 = s0[r] * SCALE_LOG2E;
            const float t1 = s1[r] * SCALE_LOG2E;
            float mx = fmaxf(t0, t1);
            mx = fmaxf(mx, __shfl_xor(mx, 1, 32));
            mx = fmaxf(mx, __shfl_xor(mx, 2, 32));
            mx = fmaxf(mx, __shfl_xor(mx, 4, 32));
            mx = fmaxf(mx, __shfl_xor(mx, 8, 32));
            const float mnew = fmaxf(mrow[r], mx);
            const float corr = fast_exp2(mrow[r] - mnew);
            const float p0   = fast_exp2(t0 - mnew);
            const float p1   = fast_exp2(t1 - mnew);
            float rs = p0 + p1;
            rs += __shfl_xor(rs, 1, 32);
            rs += __shfl_xor(rs, 2, 32);
            rs += __shfl_xor(rs, 4, 32);
            rs += __shfl_xor(rs, 8, 32);
            lrow[r] = lrow[r] * corr + rs;
            mrow[r] = mnew;
            #pragma unroll
            for (int t = 0; t < 4; ++t) accO[t][r] *= corr;
            // D-layout -> LDS (row = lh*8+r, cols ln / 16+ln)
            const int rowm = lh * 8 + r;
            pl[rowm * 32 + ln]      = (_Float16)p0;
            pl[rowm * 32 + 16 + ln] = (_Float16)p1;
        }
        asm volatile("s_wait_dscnt 0" ::: "memory");   // P stores visible wave-wide

        // ---- P as A-layout 16x32 fragment ----
        const half16 pf = *(const half16*)(pl + ln * 32 + lh * 16);

        // ---- O += P @ V (4 WMMAs over the 64 head dims), V from LDS ----
        const _Float16* vt = vTile[buf];
        #pragma unroll
        for (int t = 0; t < 4; ++t) {
            const half16 vfr = *(const half16*)(vt + (t * 16 + ln) * 32 + lh * 16);
            accO[t] = wmma_f16(pf, vfr, accO[t]);
        }
    }

    // ---- Finalize: divide by l, store f16 to O [B*N, H*DH] ----
    const int b = bh >> 4, h = bh & 15;
    #pragma unroll
    for (int t = 0; t < 4; ++t) {
        #pragma unroll
        for (int r = 0; r < 8; ++r) {
            const int n = mtile * 16 + lh * 8 + r;
            const float val = accO[t][r] / lrow[r];
            obuf[((size_t)b * NN + n) * (HH * DH) + h * DH + t * 16 + ln] = (_Float16)val;
        }
    }
}

// ---------------------------------------------------------------------------
// 5) Output projection: O[4096,1024] f16 @ Wout -> out fp32
//    (32x64 per wave, ping-pong pipelined K loop, 2 waves/EU)
// ---------------------------------------------------------------------------
__global__ __launch_bounds__(256, 2) void gemm_out_kernel(const _Float16* __restrict__ obuf,
                                                          const _Float16* __restrict__ woutT,  // [1024,1024]
                                                          float* __restrict__ out) {
    const int lane = threadIdx.x & 31;
    const int wid  = blockIdx.x * 8 + (threadIdx.x >> 5);
    const int mIdx = wid & 127;          // 128 M-tiles of 32 rows
    const int cIdx = wid >> 7;           // 16 column tiles of 64
    const int lh   = lane >> 4;
    const int ln   = lane & 15;
    const int m0   = mIdx * 32;
    const int c0   = cIdx * 64;

    const _Float16* wrowp[4];
    #pragma unroll
    for (int t = 0; t < 4; ++t) wrowp[t] = woutT + (size_t)(c0 + t * 16 + ln) * DD;
    const _Float16* arow[2] = { obuf + (size_t)(m0 + ln) * DD,
                                obuf + (size_t)(m0 + 16 + ln) * DD };

    f32x8 acc[2][4] = {};
    half16 aA[2], bA[4], aB[2], bB[4];

    auto loadA = [&](int k) {
        #pragma unroll
        for (int rb = 0; rb < 2; ++rb)
            aA[rb] = *(const half16*)(arow[rb] + k + lh * 16);
        #pragma unroll
        for (int t = 0; t < 4; ++t) bA[t] = *(const half16*)(wrowp[t] + k + lh * 16);
    };
    auto loadB = [&](int k) {
        #pragma unroll
        for (int rb = 0; rb < 2; ++rb)
            aB[rb] = *(const half16*)(arow[rb] + k + lh * 16);
        #pragma unroll
        for (int t = 0; t < 4; ++t) bB[t] = *(const half16*)(wrowp[t] + k + lh * 16);
    };
    auto compA = [&]() {
        #pragma unroll
        for (int rb = 0; rb < 2; ++rb)
            #pragma unroll
            for (int t = 0; t < 4; ++t) acc[rb][t] = wmma_f16(aA[rb], bA[t], acc[rb][t]);
    };
    auto compB = [&]() {
        #pragma unroll
        for (int rb = 0; rb < 2; ++rb)
            #pragma unroll
            for (int t = 0; t < 4; ++t) acc[rb][t] = wmma_f16(aB[rb], bB[t], acc[rb][t]);
    };

    loadA(0);
    for (int k = 0; k < DD - 64; k += 64) {
        loadB(k + 32);
        compA();
        loadA(k + 64);
        compB();
    }
    loadB(DD - 32);
    compA();
    compB();

    #pragma unroll
    for (int rb = 0; rb < 2; ++rb)
        #pragma unroll
        for (int t = 0; t < 4; ++t)
            #pragma unroll
            for (int r = 0; r < 8; ++r) {
                const int m = m0 + rb * 16 + lh * 8 + r;
                out[(size_t)m * DD + c0 + t * 16 + ln] = acc[rb][t][r];
            }
}

// ---------------------------------------------------------------------------
// Launcher
// ---------------------------------------------------------------------------
extern "C" void kernel_launch(void* const* d_in, const int* in_sizes, int n_in,
                              void* d_out, int out_size, void* d_ws, size_t ws_size,
                              hipStream_t stream) {
    (void)in_sizes; (void)n_in; (void)out_size; (void)ws_size;
    const float* x    = (const float*)d_in[0];
    const float* ln_g = (const float*)d_in[1];
    const float* ln_b = (const float*)d_in[2];
    const float* Wq   = (const float*)d_in[3];
    const float* Wkv  = (const float*)d_in[4];
    const float* Wout = (const float*)d_in[5];

    char* ws = (char*)d_ws;
    const size_t MB = 1024 * 1024;
    _Float16* xn    = (_Float16*)(ws);               //  8 MB: [4096,1024]
    _Float16* wqT   = (_Float16*)(ws + 8  * MB);     //  2 MB: [1024,1024]
    _Float16* wkvT  = (_Float16*)(ws + 10 * MB);     //  4 MB: [2048,1024]
    _Float16* woutT = (_Float16*)(ws + 14 * MB);     //  2 MB: [1024,1024]
    _Float16* qb    = (_Float16*)(ws + 16 * MB);     //  8 MB: [B,H,N,DH]
    _Float16* kb    = (_Float16*)(ws + 24 * MB);     //  8 MB: [B,H,N,DH]
    _Float16* vt    = (_Float16*)(ws + 32 * MB);     //  8 MB: [B,H,DH,N]
    _Float16* ob    = (_Float16*)(ws + 40 * MB);     //  8 MB: [4096,1024]

    wconvT_kernel<<<4096, 256, 0, stream>>>(Wq,   wqT,   1024, 1024);
    wconvT_kernel<<<8192, 256, 0, stream>>>(Wkv,  wkvT,  1024, 2048);
    wconvT_kernel<<<4096, 256, 0, stream>>>(Wout, woutT, 1024, 1024);

    layernorm_kernel<<<ROWS, 256, 0, stream>>>(x, ln_g, ln_b, xn);

    // 6144 waves = 128 M-tiles x 48 col-tiles, 8 waves per block
    gemm_qkv_kernel<<<768, 256, 0, stream>>>(xn, wqT, wkvT, qb, kb, vt);

    // 4096 waves = B*H * (N/16)
    attn_kernel<<<512, 256, 0, stream>>>(qb, kb, vt, ob);

    // 2048 waves = 128 M-tiles x 16 col-tiles
    gemm_out_kernel<<<256, 256, 0, stream>>>(ob, woutT, (float*)d_out);
}
